// RisingTideAttention_18339510354281
// MI455X (gfx1250) — matrix operand order repaired and checked by reading, not compiled, not run on hardware
//
#include <hip/hip_runtime.h>

typedef _Float16 v16h __attribute__((ext_vector_type(16)));
typedef _Float16 v8h  __attribute__((ext_vector_type(8)));
typedef _Float16 v4h  __attribute__((ext_vector_type(4)));
typedef float    v8f  __attribute__((ext_vector_type(8)));

#define DDIM 512
#define NNEUR 4096
#define MROWS 16384
#define MT 64      // rows per block
#define NT 32      // neurons per main-loop iteration

static __device__ __forceinline__ v8f v8f_zero() {
    v8f z;
#pragma unroll
    for (int i = 0; i < 8; ++i) z[i] = 0.0f;
    return z;
}

// ---------- prep: f32 row -> f16 row + row sum-of-squares (one wave per row) ----------
__global__ void rowcvt_kernel(const float* __restrict__ src, _Float16* __restrict__ dst,
                              float* __restrict__ sq, int nrows) {
    int row  = blockIdx.x * 8 + (threadIdx.x >> 5);
    int lane = threadIdx.x & 31;
    if (row >= nrows) return;
    const float* s = src + (size_t)row * DDIM;
    _Float16*    d = dst + (size_t)row * DDIM;
    float ss = 0.0f;
#pragma unroll
    for (int k = 0; k < 4; ++k) {
        int c = lane * 4 + k * 128;
        float4 f = *(const float4*)(s + c);
        ss += f.x * f.x + f.y * f.y + f.z * f.z + f.w * f.w;
        v4h h;
        h[0] = (_Float16)f.x; h[1] = (_Float16)f.y;
        h[2] = (_Float16)f.z; h[3] = (_Float16)f.w;
        *(v4h*)(d + c) = h;
    }
#pragma unroll
    for (int off = 16; off > 0; off >>= 1) ss += __shfl_xor(ss, off, 32);
    if (lane == 0) sq[row] = ss;
}

// ---------- prep: values [N, D] f32 -> valuesT [D, N] f16 (tiled transpose) ----------
__global__ void transpose_vals_kernel(const float* __restrict__ vals,
                                      _Float16* __restrict__ vT) {
    __shared__ _Float16 tile[32][33];
    int n0 = blockIdx.x * 32, d0 = blockIdx.y * 32;
    int tx = threadIdx.x, ty = threadIdx.y;  // 32 x 8
#pragma unroll
    for (int i = 0; i < 32; i += 8)
        tile[ty + i][tx] = (_Float16)vals[(size_t)(n0 + ty + i) * DDIM + d0 + tx];
    __syncthreads();
#pragma unroll
    for (int i = 0; i < 32; i += 8)
        vT[(size_t)(d0 + ty + i) * NNEUR + n0 + tx] = tile[tx][ty + i];
}

// ---------- prep: inverse effective temperature ----------
__global__ void einv_kernel(const float* __restrict__ t, const float* __restrict__ s,
                            float* __restrict__ einv, int n) {
    int i = blockIdx.x * 256 + threadIdx.x;
    if (i < n) einv[i] = 1.0f / ((fabsf(t[i]) + 0.1f) * s[i]);
}

// ---------- fused main kernel: GEMM1 -> exp weights -> GEMM2 (+rowsum via WMMA) ----------
__global__ __launch_bounds__(256, 1)
void rbf_fused_kernel(const _Float16* __restrict__ xh,   // [M, D] f16
                      const _Float16* __restrict__ ph,   // [N, D] f16
                      const _Float16* __restrict__ vT,   // [D, N] f16
                      const float* __restrict__ xsq,     // [M]
                      const float* __restrict__ psq,     // [N]
                      const float* __restrict__ einv,    // [N]
                      float* __restrict__ out) {         // [M, D] f32
    __shared__ _Float16 wlds[MT * NT];  // weights tile, row-major, 32 halfs per row

    const int lane  = threadIdx.x & 31;
    const int wid   = threadIdx.x >> 5;   // 0..7
    const int mtile = wid >> 1;           // 0..3 : which 16-row group
    const int sel   = wid & 1;            // GEMM1 n-half / GEMM2 column-half
    const int lo    = lane & 15;
    const int hi    = lane >> 4;          // 0/1
    const int mbase = blockIdx.x * MT + mtile * 16;

    // Cache x A-fragments for this wave's 16 rows: K = 512 in 16 chunks of 32.
    v16h afrag[16];
    {
        const _Float16* xrow = xh + (size_t)(mbase + lo) * DDIM;
#pragma unroll
        for (int kc = 0; kc < 16; ++kc) {
            int k0 = kc * 32 + hi * 16;
            v8h a0 = *(const v8h*)(xrow + k0);
            v8h a1 = *(const v8h*)(xrow + k0 + 8);
            afrag[kc] = __builtin_shufflevector(a0, a1, 0, 1, 2, 3, 4, 5, 6, 7,
                                                8, 9, 10, 11, 12, 13, 14, 15);
        }
    }

    // Per-lane ||x||^2 matching the WMMA C layout (M = v + 8*hi).
    float xsqr[8];
#pragma unroll
    for (int v = 0; v < 8; ++v) xsqr[v] = xsq[mbase + v + 8 * hi];

    // Output accumulators: 16 col-tiles of 16 -> 256 columns starting at sel*256.
    v8f acc[16];
#pragma unroll
    for (int i = 0; i < 16; ++i) acc[i] = v8f_zero();
    v8f rsum = v8f_zero();

    v16h ones;
#pragma unroll
    for (int i = 0; i < 16; ++i) ones[i] = (_Float16)1.0f;

    const int cb = sel * 256;

    for (int nb = 0; nb < NNEUR; nb += NT) {
        const int nbw = nb + sel * 16;  // this wave's 16 neurons for GEMM1

        // ---- GEMM1: cross[16x16] = x_tile @ positions_tile^T ----
        v8f cross = v8f_zero();
        const _Float16* prow = ph + (size_t)(nbw + lo) * DDIM + hi * 16;
#pragma unroll
        for (int kc = 0; kc < 16; ++kc) {
            v16h b = *(const v16h*)(prow + kc * 32);
            cross = __builtin_amdgcn_wmma_f32_16x16x32_f16(
                false, afrag[kc], false, b, (short)0, cross, false, false);
        }

        const float psqr  = psq[nbw + lo];
        const float einvr = einv[nbw + lo];

        __syncthreads();  // previous iteration's LDS reads are complete
#pragma unroll
        for (int v = 0; v < 8; ++v) {
            float d2 = fmaxf(xsqr[v] + psqr - 2.0f * cross[v], 0.0f);
            float w  = __expf(-__fsqrt_rn(d2) * einvr);
            int row = mtile * 16 + v + 8 * hi;
            wlds[row * NT + sel * 16 + lo] = (_Float16)w;
        }
        __syncthreads();  // full 64x32 weights tile visible

        // ---- weights A-fragment for this wave's GEMM2 m-tile (all 32 n) ----
        const _Float16* wrow = wlds + (mtile * 16 + lo) * NT + hi * 16;
        v8h w0 = *(const v8h*)(wrow);
        v8h w1 = *(const v8h*)(wrow + 8);
        v16h wa = __builtin_shufflevector(w0, w1, 0, 1, 2, 3, 4, 5, 6, 7,
                                          8, 9, 10, 11, 12, 13, 14, 15);

        // Row sums via ones-matrix WMMA (same C layout as acc).
        rsum = __builtin_amdgcn_wmma_f32_16x16x32_f16(
            false, wa, false, ones, (short)0, rsum, false, false);

        // ---- GEMM2: acc[16x256] += weights @ values ----
        const _Float16* vcol = vT + (size_t)(cb + lo) * NNEUR + nb + hi * 16;
#pragma unroll
        for (int ct = 0; ct < 16; ++ct) {
            v16h b = *(const v16h*)(vcol + (size_t)ct * 16 * NNEUR);
            acc[ct] = __builtin_amdgcn_wmma_f32_16x16x32_f16(
                false, wa, false, b, (short)0, acc[ct], false, false);
        }
    }

    // ---- normalize by row sums and store ----
    float scale[8];
#pragma unroll
    for (int v = 0; v < 8; ++v) scale[v] = 1.0f / (rsum[v] + 1e-8f);
#pragma unroll
    for (int ct = 0; ct < 16; ++ct) {
#pragma unroll
        for (int v = 0; v < 8; ++v) {
            out[(size_t)(mbase + v + 8 * hi) * DDIM + cb + ct * 16 + lo] =
                acc[ct][v] * scale[v];
        }
    }
}

extern "C" void kernel_launch(void* const* d_in, const int* in_sizes, int n_in,
                              void* d_out, int out_size, void* d_ws, size_t ws_size,
                              hipStream_t stream) {
    const float* x      = (const float*)d_in[0];  // [8,2048,512]
    const float* pos    = (const float*)d_in[1];  // [4096,512]
    const float* vals   = (const float*)d_in[2];  // [4096,512]
    const float* temp   = (const float*)d_in[3];  // [4096]
    const float* nscale = (const float*)d_in[4];  // [4096]
    float* out = (float*)d_out;

    // workspace layout (~25.3 MB)
    char* ws = (char*)d_ws;
    _Float16* xh  = (_Float16*)(ws);                               // 16 MiB
    _Float16* ph  = (_Float16*)(ws + (16u << 20));                 //  4 MiB
    _Float16* vT  = (_Float16*)(ws + (20u << 20));                 //  4 MiB
    float* xsq    = (float*)(ws + (24u << 20));                    // 64 KiB
    float* psq    = (float*)(ws + (24u << 20) + (64u << 10));      // 16 KiB
    float* einv   = (float*)(ws + (24u << 20) + (80u << 10));      // 16 KiB

    rowcvt_kernel<<<MROWS / 8, 256, 0, stream>>>(x, xh, xsq, MROWS);
    rowcvt_kernel<<<NNEUR / 8, 256, 0, stream>>>(pos, ph, psq, NNEUR);
    transpose_vals_kernel<<<dim3(NNEUR / 32, DDIM / 32), dim3(32, 8), 0, stream>>>(vals, vT);
    einv_kernel<<<NNEUR / 256, 256, 0, stream>>>(temp, nscale, einv, NNEUR);

    rbf_fused_kernel<<<MROWS / MT, 256, 0, stream>>>(xh, ph, vT, xsq, psq, einv, out);
}
